// MultiHeadAttention_5695126634594
// MI455X (gfx1250) — matrix operand order repaired and checked by reading, not compiled, not run on hardware
//
#include <hip/hip_runtime.h>
#include <hip/hip_bf16.h>
#include <math.h>

typedef __attribute__((ext_vector_type(16))) __bf16 v16bf;
typedef __attribute__((ext_vector_type(8)))  float  v8f;
typedef __attribute__((ext_vector_type(4)))  float  f32x4;
typedef unsigned short u16;
typedef __attribute__((ext_vector_type(8)))  u16    u16x8;
typedef __attribute__((ext_vector_type(16))) u16    u16x16;

// Problem constants (reference: B,N,M,D,H = 2,2048,2048,1024,16)
constexpr int Bc = 2, Nc = 2048, Mc = 2048, Dc = 1024, Hc = 16, DHc = 64;
constexpr int ROWS = Bc * Nc;   // 4096 token rows per projection

// Native bf16 convert (lowers to v_cvt hardware ops on gfx1250, RNE)
__device__ __forceinline__ u16 f2bf_bits(float x) {
  return __builtin_bit_cast(u16, (__bf16)x);
}
// Assemble a WMMA A/B operand from 4 float4s (16 fp32 -> 16 bf16)
__device__ __forceinline__ v16bf cvt16(f32x4 a0, f32x4 a1, f32x4 a2, f32x4 a3) {
  v16bf r;
  for (int i = 0; i < 4; ++i) {
    r[i]      = (__bf16)a0[i];
    r[4 + i]  = (__bf16)a1[i];
    r[8 + i]  = (__bf16)a2[i];
    r[12 + i] = (__bf16)a3[i];
  }
  return r;
}
__device__ __forceinline__ v16bf bf16v(u16x8 lo, u16x8 hi) {
  u16x16 u;
  for (int i = 0; i < 8; ++i) { u[i] = lo[i]; u[8 + i] = hi[i]; }
  return __builtin_bit_cast(v16bf, u);
}

// ---------------------------------------------------------------------------
// Kernel 1: projection  out = tok @ W^T + bias   (fp32 in, bf16 out)
// Each wave: 16 rows x 64 cols, K-loop over D in steps of 32.
// transposed=0: out[row][j] row-major (for q,k)
// transposed=1: out = vT[b][h][c][m]  (for v; makes AV B-operand contiguous)
// ---------------------------------------------------------------------------
__global__ __launch_bounds__(256) void proj_kernel(
    const float* __restrict__ tok, const float* __restrict__ W,
    const float* __restrict__ bias, u16* __restrict__ out, int transposed)
{
  const int lane = threadIdx.x & 31;
  const int g    = lane >> 4;       // half-wave group
  const int l16  = lane & 15;
  const int wid  = blockIdx.x * (blockDim.x >> 5) + (threadIdx.x >> 5);
  const int rowBase = (wid >> 4) * 16;      // 256 row tiles
  const int colBase = (wid & 15) * 64;      // 16 col groups of 64

  v8f acc[4] = {};
  const float* aptr = tok + (size_t)(rowBase + l16) * Dc;

  for (int kk0 = 0; kk0 < Dc; kk0 += 32) {
    // A tile: lane row = rowBase+l16; elems 0-15 -> k=kk0+g*16+e
    const float* p = aptr + kk0 + g * 16;
    v16bf a = cvt16(*(const f32x4*)(p),      *(const f32x4*)(p + 4),
                    *(const f32x4*)(p + 8), *(const f32x4*)(p + 12));
    for (int t = 0; t < 4; ++t) {
      // B tile: col j fixed per lane; k = kk0 + g*16 + e contiguous in W row j
      const int j = colBase + t * 16 + l16;
      const float* q = W + (size_t)j * Dc + kk0 + g * 16;
      v16bf bmat = cvt16(*(const f32x4*)(q),     *(const f32x4*)(q + 4),
                         *(const f32x4*)(q + 8), *(const f32x4*)(q + 12));
      acc[t] = __builtin_amdgcn_wmma_f32_16x16x32_bf16(
          false, a, false, bmat, (short)0, acc[t], false, false);
    }
  }

  for (int t = 0; t < 4; ++t) {
    const int j  = colBase + t * 16 + l16;
    const float bv = bias[j];
    if (!transposed) {
      for (int r = 0; r < 8; ++r) {
        const int row = rowBase + g * 8 + r;    // C/D: row = g*8 + r
        out[(size_t)row * Dc + j] = f2bf_bits(acc[t][r] + bv);
      }
    } else {
      // vT[b][h][c][m]: consecutive r are consecutive m -> one 128-bit store
      const int bb    = rowBase / Mc;
      const int mBase = (rowBase % Mc) + g * 8;
      const int h = j >> 6, c = j & 63;
      u16x8 pk;
      for (int r = 0; r < 8; ++r) pk[r] = f2bf_bits(acc[t][r] + bv);
      *(u16x8*)(out + ((size_t)(bb * Hc + h) * DHc + c) * Mc + mBase) = pk;
    }
  }
}

// ---------------------------------------------------------------------------
// Kernel 2 (fused): scores + softmax + attn@v for one (b, h, 16-row n-tile).
// 256 threads = 8 waves. Score strip S[16][2048] fp32 lives in LDS (128 KB);
// attn makes exactly ONE trip to HBM (NT store), and the AV GEMM consumes
// the strip straight from LDS. qkw is streamed with NT loads so the hot
// q/k/vT bf16 operands (24 MB) stay resident in the 192 MB L2.
// ---------------------------------------------------------------------------
__global__ __launch_bounds__(256) void fused_attn_kernel(
    const u16* __restrict__ qbf, const u16* __restrict__ kbf,
    const u16* __restrict__ vT,  const float* __restrict__ qkw,
    const unsigned char* __restrict__ kmask,
    float* __restrict__ attn, float* __restrict__ hidden)
{
  __shared__ float S[16 * Mc];      // 128 KB score strip
  __shared__ float red[256];        // softmax reduction scratch
  __shared__ float hbuf[16 * DHc];  // 4 KB AV partial-sum tile

  const int tid  = threadIdx.x;
  const int lane = tid & 31;
  const int g    = lane >> 4;
  const int l16  = lane & 15;
  const int w    = tid >> 5;                     // wave id 0..7

  const int bh      = blockIdx.x / (Nc / 16);
  const int nTile   = blockIdx.x % (Nc / 16);
  const int b       = bh / Hc, h = bh % Hc;
  const int rowBase = nTile * 16;

  // zero the AV accumulation tile (1024 floats)
  for (int i = tid; i < 16 * DHc; i += 256) hbuf[i] = 0.0f;

  // ---------------- Phase 1: scores -> LDS ----------------
  const u16* qrow = qbf + (size_t)(b * Nc + rowBase + l16) * Dc + h * DHc;
  const float scale = 0.125f;   // 1/sqrt(64)

  for (int cg = 0; cg < 4; ++cg) {           // wave covers m in [w*256, w*256+256)
    const int colBase = w * 256 + cg * 64;
    v8f acc[4] = {};
    for (int kk0 = 0; kk0 < DHc; kk0 += 32) {
      const u16* ap = qrow + kk0 + g * 16;
      v16bf a = bf16v(*(const u16x8*)ap, *(const u16x8*)(ap + 8));
      for (int t = 0; t < 4; ++t) {
        const int m = colBase + t * 16 + l16;
        const u16* p = kbf + (size_t)(b * Mc + m) * Dc + h * DHc + kk0 + g * 16;
        v16bf bmat = bf16v(*(const u16x8*)p, *(const u16x8*)(p + 8));
        acc[t] = __builtin_amdgcn_wmma_f32_16x16x32_bf16(
            false, a, false, bmat, (short)0, acc[t], false, false);
      }
    }
    for (int t = 0; t < 4; ++t) {
      const int m = colBase + t * 16 + l16;
      const bool masked = kmask[b * Mc + m] != 0;
      for (int r = 0; r < 8; ++r) {
        const int nr = g * 8 + r;        // row within tile
        const float wgt = __builtin_nontemporal_load(
            qkw + ((size_t)b * Nc + rowBase + nr) * Mc + m);
        float s = acc[t][r] * scale * wgt;
        if (masked) s = -__builtin_inff();
        S[nr * Mc + m] = s;
      }
    }
  }
  __syncthreads();

  // ---------------- Phase 2: softmax in LDS + single attn write ------------
  // 16 threads per row, 128 contiguous elements per thread.
  const int srow = tid >> 4;
  const int seg  = (tid & 15) * 128;
  float* sp = &S[srow * Mc + seg];

  float lmax = -__builtin_inff();
  for (int i = 0; i < 128; i += 4) {
    f32x4 v = *(const f32x4*)(sp + i);
    lmax = fmaxf(lmax, fmaxf(fmaxf(v[0], v[1]), fmaxf(v[2], v[3])));
  }
  red[tid] = lmax;
  __syncthreads();
  float mx = -__builtin_inff();
  for (int i = 0; i < 16; ++i) mx = fmaxf(mx, red[(srow << 4) + i]);
  __syncthreads();

  float lsum = 0.0f;
  for (int i = 0; i < 128; i += 4) {
    f32x4 v = *(const f32x4*)(sp + i);
    f32x4 e;
    for (int j = 0; j < 4; ++j) { e[j] = __expf(v[j] - mx); lsum += e[j]; }
    *(f32x4*)(sp + i) = e;
  }
  red[tid] = lsum;
  __syncthreads();
  float rsum = 0.0f;
  for (int i = 0; i < 16; ++i) rsum += red[(srow << 4) + i];
  const float inv = 1.0f / rsum;

  // normalize in LDS and stream the final attn row segment to HBM (NT, only pass)
  float* arow = attn + ((size_t)bh * Nc + rowBase + srow) * Mc + seg;
  for (int i = 0; i < 128; i += 4) {
    f32x4 e = *(const f32x4*)(sp + i);
    f32x4 o;
    for (int j = 0; j < 4; ++j) o[j] = e[j] * inv;
    *(f32x4*)(sp + i) = o;
    __builtin_nontemporal_store(o, (f32x4*)(arow + i));
  }
  __syncthreads();

  // ---------------- Phase 3: AV from LDS; K split across waves -------------
  // Wave w handles m in [w*256, (w+1)*256); partial 16x64 summed via ds_add.
  {
    v8f acc[4] = {};
    for (int kk = 0; kk < 256; kk += 32) {
      const int kk0 = w * 256 + kk;
      const float* p = &S[l16 * Mc + kk0 + g * 16];     // A row = l16
      v16bf a = cvt16(*(const f32x4*)(p),      *(const f32x4*)(p + 4),
                      *(const f32x4*)(p + 8), *(const f32x4*)(p + 12));
      for (int t = 0; t < 4; ++t) {
        const int c = t * 16 + l16;
        const u16* q = vT + ((size_t)bh * DHc + c) * Mc + kk0 + g * 16;
        v16bf bmat = bf16v(*(const u16x8*)q, *(const u16x8*)(q + 8));
        acc[t] = __builtin_amdgcn_wmma_f32_16x16x32_bf16(
            false, a, false, bmat, (short)0, acc[t], false, false);
      }
    }
    for (int t = 0; t < 4; ++t) {
      const int c = t * 16 + l16;
      for (int r = 0; r < 8; ++r)
        atomicAdd(&hbuf[(g * 8 + r) * DHc + c], acc[t][r]);
    }
  }
  __syncthreads();

  // ---------------- Phase 4: write hidden tile -----------------------------
  // 1024 floats; each thread writes one aligned float4 run.
  {
    const int e = tid * 4;                 // 0..1020
    const int nr = e / DHc, c = e % DHc;
    f32x4 v = *(const f32x4*)(&hbuf[e]);
    *(f32x4*)(hidden + ((size_t)b * Nc + rowBase + nr) * Dc + h * DHc + c) = v;
  }
}

// ---------------------------------------------------------------------------
extern "C" void kernel_launch(void* const* d_in, const int* in_sizes, int n_in,
                              void* d_out, int out_size, void* d_ws, size_t ws_size,
                              hipStream_t stream)
{
  const float* q_tok = (const float*)d_in[0];
  const float* k_tok = (const float*)d_in[1];
  const float* v_tok = (const float*)d_in[2];
  const float* qkw   = (const float*)d_in[3];
  const unsigned char* kmask = (const unsigned char*)d_in[4];  // numpy bool = 1 byte
  const float* Wq = (const float*)d_in[5]; const float* bq = (const float*)d_in[6];
  const float* Wk = (const float*)d_in[7]; const float* bk = (const float*)d_in[8];
  const float* Wv = (const float*)d_in[9]; const float* bv = (const float*)d_in[10];

  float* hidden = (float*)d_out;                        // (B,N,D)
  float* attn   = hidden + (size_t)Bc * Nc * Dc;        // (B,H,N,M)

  u16* qbf = (u16*)d_ws;                                // 8 MB
  u16* kbf = qbf + (size_t)ROWS * Dc;                   // 8 MB
  u16* vT  = kbf + (size_t)ROWS * Dc;                   // 8 MB (b,h,c,m)

  dim3 blk(256);
  // 4096 waves per projection, 8 waves/block
  proj_kernel<<<512, blk, 0, stream>>>(q_tok, Wq, bq, qbf, 0);
  proj_kernel<<<512, blk, 0, stream>>>(k_tok, Wk, bk, kbf, 0);
  proj_kernel<<<512, blk, 0, stream>>>(v_tok, Wv, bv, vT, 1);
  // one block per (b, h, 16-row n-tile): 2*16*128 = 4096 blocks
  fused_attn_kernel<<<Bc * Hc * (Nc / 16), blk, 0, stream>>>(
      qbf, kbf, vT, qkw, kmask, attn, hidden);
}
